// GatedLinearRNN_75350906241524
// MI455X (gfx1250) — matrix operand order repaired and checked by reading, not compiled, not run on hardware
//
#include <hip/hip_runtime.h>

// ---------------------------------------------------------------------------
// GatedLinearRNN on MI455X (gfx1250):
//   u = x@Wp + bp ; inp=tanh(u0) ; ig=sig(u1) ; og=sig(u2)
//   h_t = (1-ig_t) h_{t-1} + inp_t*ig_t   (length-4096 scan, chunked 3-pass)
//   y = (tanh(h)*og) @ Wo + bo ; outputs (h_last, y)
// bf16 WMMA GEMMs; B panels staged into double-buffered LDS by the Tensor
// Data Mover (tensor_load_to_lds, TENSORcnt), A fragments double-buffered in
// registers, all B fragments preloaded per step so the WMMA chain issues
// back-to-back, activations fused into the GEMM1 epilogue.
// ---------------------------------------------------------------------------

typedef __attribute__((ext_vector_type(16))) __bf16 bf16x16;
typedef __attribute__((ext_vector_type(8)))  __bf16 bf16x8;
typedef __attribute__((ext_vector_type(8)))  float  f32x8;
typedef __attribute__((ext_vector_type(4)))  unsigned int u32x4;
typedef __attribute__((ext_vector_type(4)))  int i32x4;
typedef __attribute__((ext_vector_type(8)))  int i32x8;

#define DEV static __device__ __forceinline__

constexpr int Bn    = 4;
constexpr int S     = 4096;
constexpr int DM    = 1024;
constexpr int DH    = 1024;
constexpr int M     = Bn * S;        // 16384 rows
constexpr int NCH   = 16;            // scan chunks along S
constexpr int CHUNK = S / NCH;       // 256

// fp32 -> bf16, round-to-nearest-even
DEV __bf16 f2bf(float f) {
  unsigned u = __builtin_bit_cast(unsigned, f);
  unsigned r = (u + 0x7FFFu + ((u >> 16) & 1u)) >> 16;
  unsigned short h = (unsigned short)r;
  return __builtin_bit_cast(__bf16, h);
}

DEV float sigmoidf(float x) { return 1.0f / (1.0f + __expf(-x)); }

// ---------------------------------------------------------------------------
// Tensor Data Mover: DMA a (tile_z, tile_y, tile_x) bf16 tile (strides in
// elements) from global into LDS at lds_byte_off, packed contiguously
// z-major/y/x-minor. Tensor dims == tile dims so no OOB path is exercised.
// D# layout per CDNA5 ISA 8.3-8.6 (count=1, type=2 "image", data_size=2B,
// workgroup_mask=0: not in a cluster).
DEV void tdm_load_tile(const __bf16* gsrc, unsigned lds_byte_off, int tile_x,
                       int tile_y, int tile_z, long long stride_y,
                       long long stride_z) {
  unsigned long long ga = (unsigned long long)(size_t)gsrc;
  u32x4 g0;
  g0[0] = 1u;                                        // count=1, user desc
  g0[1] = lds_byte_off;                              // lds_addr (bytes)
  g0[2] = (unsigned)(ga & 0xFFFFFFFFull);            // global_addr[31:0]
  g0[3] = (unsigned)((ga >> 32) & 0x1FFFFFFull) | (2u << 30);  // [56:32]|type=2

  unsigned long long sy = (unsigned long long)stride_y;
  unsigned long long sz = (unsigned long long)stride_z;
  i32x8 g1;
  g1[0] = (int)(1u << 16);  // workgroup_mask=0, data_size=1 (2 bytes)
  g1[1] = (int)(((unsigned)tile_x & 0xFFFFu) << 16);  // dim0[15:0] hi half
  g1[2] = (int)((((unsigned)tile_x) >> 16) |
                (((unsigned)tile_y & 0xFFFFu) << 16));  // dim0 hi | dim1 lo
  g1[3] = (int)((((unsigned)tile_y) >> 16) |
                (((unsigned)tile_x & 0xFFFFu) << 16));  // dim1 hi | tile_dim0
  g1[4] = (int)(((unsigned)tile_y & 0xFFFFu) |
                (((unsigned)tile_z & 0xFFFFu) << 16));  // tile_dim1|tile_dim2
  g1[5] = (int)(unsigned)(sy & 0xFFFFFFFFull);          // dim0_stride[31:0]
  g1[6] = (int)((unsigned)((sy >> 32) & 0xFFFFull) |
                (unsigned)((sz & 0xFFFFull) << 16));  // d0s[47:32]|d1s[15:0]
  g1[7] = (int)(unsigned)((sz >> 16) & 0xFFFFFFFFull);  // dim1_stride[47:16]

  i32x4 g2;
  g2[0] = tile_z;  // tensor_dim2
  g2[1] = 0;       // tensor_dim3 / lds_addr_increment
  g2[2] = 0;       // tensor_dim2_stride[31:0]
  g2[3] = 0;       // dim2_stride[47:32] | tile_dim3=0
  i32x4 g3 = (i32x4)0;

#if defined(__clang_major__) && (__clang_major__ >= 23)
  i32x8 gx = (i32x8)0;
  __builtin_amdgcn_tensor_load_to_lds(g0, g1, g2, g3, gx, 0);
#else
  __builtin_amdgcn_tensor_load_to_lds(g0, g1, g2, g3, 0);
#endif
}

// A-fragment (16x32 bf16, row-major source, row pitch lda).
// Lanes 0-15: rows M=0..15, K={0..7,16..23}; lanes 16-31: K={8..15,24..31}.
DEV bf16x16 load_a_frag(const __bf16* __restrict__ A, int lda, int m0, int k0,
                        int lane) {
  const int lo = lane & 15, hi = lane >> 4;
  const __bf16* p = A + (size_t)(m0 + lo) * lda + (k0 + hi * 8);
  bf16x8 p0 = *reinterpret_cast<const bf16x8*>(p);
  bf16x8 p1 = *reinterpret_cast<const bf16x8*>(p + 16);
  bf16x16 r;
#pragma unroll
  for (int i = 0; i < 8; ++i) { r[i] = p0[i]; r[8 + i] = p1[i]; }
  return r;
}

// B-fragment from an LDS tile stored row-major [K=32][N=ldn].
// Lane l holds N-col t*16+(l&15), K={0..7,16..23} (lanes 0-15) / {8..15,24..31}.
DEV bf16x16 lds_b_frag(const __bf16* base, int ldn, int t, int lane) {
  const int kh = (lane >> 4) * 8;
  const __bf16* p = base + kh * ldn + t * 16 + (lane & 15);
  bf16x16 r;
#pragma unroll
  for (int i = 0; i < 8; ++i) { r[i] = p[i * ldn]; r[8 + i] = p[(16 + i) * ldn]; }
  return r;
}

// ---------------------------------------------------------------------------
__global__ __launch_bounds__(256) void conv_f32_to_bf16(
    const float* __restrict__ src, __bf16* __restrict__ dst, int n) {
  int i = blockIdx.x * 256 + threadIdx.x;
  if (i < n) dst[i] = f2bf(src[i]);
}

// ---------------------------------------------------------------------------
// GEMM1 fused: block = 256 threads (8 waves) covering 128 rows x 32 cols,
// replicated over the three Wp column groups. B panel (3 strips of 32x32 at
// stride DH) = one 3D TDM descriptor per K-step, double-buffered in LDS.
__global__ __launch_bounds__(256) void gemm1_fused(
    const __bf16* __restrict__ xb, const __bf16* __restrict__ WpB,
    const float* __restrict__ bp, float* __restrict__ aA,
    float* __restrict__ sA, float* __restrict__ ogA) {
  __shared__ alignas(32) __bf16 lds[2][3 * 32 * 32];  // 12 KB total
  const int tid  = threadIdx.x;
  const int lane = tid & 31;
  const int wave = tid >> 5;
  const int m0 = blockIdx.y * 128 + wave * 16;
  const int n0 = blockIdx.x * 32;

  f32x8 acc[3][2];
  const f32x8 zero = {};
#pragma unroll
  for (int g = 0; g < 3; ++g)
#pragma unroll
    for (int t = 0; t < 2; ++t) acc[g][t] = zero;

  // Prologue: TDM-stage k0=0 panel, preload first A fragment.
  if (wave == 0)
    tdm_load_tile(WpB + n0, (unsigned)(size_t)&lds[0][0], 32, 32, 3,
                  3 * DH, DH);
  bf16x16 af = load_a_frag(xb, DM, m0, 0, lane);
  if (wave == 0) __builtin_amdgcn_s_wait_tensorcnt(0);
  __syncthreads();

  int buf = 0;
  for (int k0 = 0; k0 < DM; k0 += 32) {
    const bool more = (k0 + 32) < DM;
    bf16x16 af_n;
    if (more) {
      if (wave == 0)
        tdm_load_tile(WpB + (size_t)(k0 + 32) * (3 * DH) + n0,
                      (unsigned)(size_t)&lds[buf ^ 1][0], 32, 32, 3,
                      3 * DH, DH);
      af_n = load_a_frag(xb, DM, m0, k0 + 32, lane);
    }
    // Preload all 6 B fragments, then an uninterrupted WMMA chain.
    bf16x16 bfr[6];
#pragma unroll
    for (int g = 0; g < 3; ++g)
#pragma unroll
      for (int t = 0; t < 2; ++t)
        bfr[g * 2 + t] = lds_b_frag(&lds[buf][g * 1024], 32, t, lane);
#pragma unroll
    for (int g = 0; g < 3; ++g)
#pragma unroll
      for (int t = 0; t < 2; ++t)
        acc[g][t] = __builtin_amdgcn_wmma_f32_16x16x32_bf16(
            false, af, false, bfr[g * 2 + t], (short)0, acc[g][t], false,
            false);
    if (more) {
      if (wave == 0) __builtin_amdgcn_s_wait_tensorcnt(0);
      __syncthreads();
      buf ^= 1;
      af = af_n;
    }
  }

  const int lo = lane & 15, hi = lane >> 4;
#pragma unroll
  for (int t = 0; t < 2; ++t) {
    const int j = n0 + t * 16 + lo;
    const float b0 = bp[j], b1 = bp[DH + j], b2 = bp[2 * DH + j];
#pragma unroll
    for (int r = 0; r < 8; ++r) {
      const int mg = m0 + r + hi * 8;  // global row in [0, B*S)
      float inp = tanhf(acc[0][t][r] + b0);
      float ig  = sigmoidf(acc[1][t][r] + b1);
      float og  = sigmoidf(acc[2][t][r] + b2);
      size_t idx = (size_t)mg * DH + j;
      aA[idx]  = 1.0f - ig;
      sA[idx]  = inp * ig;
      ogA[idx] = og;
    }
  }
}

// ---------------------------------------------------------------------------
// Chunked scan passes.
__global__ __launch_bounds__(256) void scan_pass1(
    const float* __restrict__ aA, const float* __restrict__ sA,
    float* __restrict__ Ac, float* __restrict__ Hl) {
  int tid = blockIdx.x * 256 + threadIdx.x;       // B*NCH*DH threads
  int j = tid & (DH - 1);
  int c = (tid >> 10) & (NCH - 1);
  int b = tid >> 14;
  size_t base = (size_t)(b * S + c * CHUNK) * DH + j;
  float p = 1.0f, h = 0.0f;
  for (int i = 0; i < CHUNK; ++i) {
    float av = aA[base], sv = sA[base];
    p *= av;
    h = av * h + sv;
    base += DH;
  }
  int idx = (b * NCH + c) * DH + j;
  Ac[idx] = p;
  Hl[idx] = h;
}

__global__ __launch_bounds__(256) void scan_pass2(
    const float* __restrict__ Ac, const float* __restrict__ Hl,
    float* __restrict__ Hp, float* __restrict__ hlast) {
  int tid = blockIdx.x * 256 + threadIdx.x;       // B*DH threads
  int j = tid & (DH - 1);
  int b = tid >> 10;
  float h = 0.0f;
#pragma unroll
  for (int c = 0; c < NCH; ++c) {
    int idx = (b * NCH + c) * DH + j;
    Hp[idx] = h;                                   // prefix entering chunk c
    h = Ac[idx] * h + Hl[idx];
  }
  hlast[b * DH + j] = h;                           // output[0]: h[:, -1, :]
}

__global__ __launch_bounds__(256) void scan_pass3(
    const float* __restrict__ aA, const float* __restrict__ sA,
    const float* __restrict__ ogA, const float* __restrict__ Hp,
    __bf16* __restrict__ ybar) {
  int tid = blockIdx.x * 256 + threadIdx.x;       // B*NCH*DH threads
  int j = tid & (DH - 1);
  int c = (tid >> 10) & (NCH - 1);
  int b = tid >> 14;
  float h = Hp[(b * NCH + c) * DH + j];
  size_t base = (size_t)(b * S + c * CHUNK) * DH + j;
  for (int i = 0; i < CHUNK; ++i) {
    float av = aA[base], sv = sA[base];
    h = av * h + sv;
    ybar[base] = f2bf(tanhf(h) * ogA[base]);
    base += DH;
  }
}

// ---------------------------------------------------------------------------
// GEMM2: y = ybar @ Wo + bo. Block = 8 waves, 128 rows x 64 cols; B panel
// (32x64) staged per K-step by a 2D TDM descriptor, double-buffered in LDS.
__global__ __launch_bounds__(256) void gemm2(
    const __bf16* __restrict__ yb, const __bf16* __restrict__ WoB,
    const float* __restrict__ bo, float* __restrict__ yout) {
  __shared__ alignas(32) __bf16 lds[2][32 * 64];  // 8 KB total
  const int tid  = threadIdx.x;
  const int lane = tid & 31;
  const int wave = tid >> 5;
  const int m0 = blockIdx.y * 128 + wave * 16;
  const int n0 = blockIdx.x * 64;

  f32x8 acc[4];
  const f32x8 zero = {};
#pragma unroll
  for (int t = 0; t < 4; ++t) acc[t] = zero;

  if (wave == 0)
    tdm_load_tile(WoB + n0, (unsigned)(size_t)&lds[0][0], 64, 32, 1, DM, 0);
  bf16x16 af = load_a_frag(yb, DH, m0, 0, lane);
  if (wave == 0) __builtin_amdgcn_s_wait_tensorcnt(0);
  __syncthreads();

  int buf = 0;
  for (int k0 = 0; k0 < DH; k0 += 32) {
    const bool more = (k0 + 32) < DH;
    bf16x16 af_n;
    if (more) {
      if (wave == 0)
        tdm_load_tile(WoB + (size_t)(k0 + 32) * DM + n0,
                      (unsigned)(size_t)&lds[buf ^ 1][0], 64, 32, 1, DM, 0);
      af_n = load_a_frag(yb, DH, m0, k0 + 32, lane);
    }
    bf16x16 bfr[4];
#pragma unroll
    for (int t = 0; t < 4; ++t)
      bfr[t] = lds_b_frag(&lds[buf][0], 64, t, lane);
#pragma unroll
    for (int t = 0; t < 4; ++t)
      acc[t] = __builtin_amdgcn_wmma_f32_16x16x32_bf16(
          false, af, false, bfr[t], (short)0, acc[t], false, false);
    if (more) {
      if (wave == 0) __builtin_amdgcn_s_wait_tensorcnt(0);
      __syncthreads();
      buf ^= 1;
      af = af_n;
    }
  }

  const int lo = lane & 15, hi = lane >> 4;
#pragma unroll
  for (int t = 0; t < 4; ++t) {
    const int n = n0 + t * 16 + lo;
    const float bb = bo[n];
#pragma unroll
    for (int r = 0; r < 8; ++r) {
      const int mg = m0 + r + hi * 8;
      yout[(size_t)mg * DM + n] = acc[t][r] + bb;
    }
  }
}

// ---------------------------------------------------------------------------
extern "C" void kernel_launch(void* const* d_in, const int* in_sizes, int n_in,
                              void* d_out, int out_size, void* d_ws,
                              size_t ws_size, hipStream_t stream) {
  const float* x  = (const float*)d_in[0];   // (B,S,DM)
  const float* Wp = (const float*)d_in[1];   // (DM, 3*DH)
  const float* bp = (const float*)d_in[2];   // (3*DH)
  const float* Wo = (const float*)d_in[3];   // (DH, DM)
  const float* bo = (const float*)d_in[4];   // (DM)

  float* out    = (float*)d_out;
  float* h_last = out;                      // B*DH floats
  float* y_out  = out + (size_t)Bn * DH;    // B*S*DM floats

  char* w = (char*)d_ws;
  auto carve = [&](size_t bytes) { char* p = w; w += bytes; return p; };
  __bf16* xb   = (__bf16*)carve((size_t)M * DM * 2);        //  32 MB
  __bf16* WpB  = (__bf16*)carve((size_t)DM * 3 * DH * 2);   //   6 MB
  __bf16* WoB  = (__bf16*)carve((size_t)DH * DM * 2);       //   2 MB
  float*  aA   = (float*)carve((size_t)M * DH * 4);         //  64 MB
  float*  sA   = (float*)carve((size_t)M * DH * 4);         //  64 MB
  float*  ogA  = (float*)carve((size_t)M * DH * 4);         //  64 MB
  __bf16* ybar = (__bf16*)carve((size_t)M * DH * 2);        //  32 MB
  float*  Ac   = (float*)carve((size_t)Bn * NCH * DH * 4);
  float*  Hl   = (float*)carve((size_t)Bn * NCH * DH * 4);
  float*  Hp   = (float*)carve((size_t)Bn * NCH * DH * 4);
  (void)ws_size; (void)in_sizes; (void)n_in; (void)out_size;

  // 1) bf16 conversions (weights + x fit comfortably in the 192MB L2).
  conv_f32_to_bf16<<<(M * DM) / 256, 256, 0, stream>>>(x, xb, M * DM);
  conv_f32_to_bf16<<<(DM * 3 * DH) / 256, 256, 0, stream>>>(Wp, WpB,
                                                            DM * 3 * DH);
  conv_f32_to_bf16<<<(DH * DM) / 256, 256, 0, stream>>>(Wo, WoB, DH * DM);

  // 2) GEMM1 + fused gate activations.
  gemm1_fused<<<dim3(DH / 32, M / 128), 256, 0, stream>>>(xb, WpB, bp, aA, sA,
                                                          ogA);

  // 3) chunked linear-recurrence scan.
  scan_pass1<<<(Bn * NCH * DH) / 256, 256, 0, stream>>>(aA, sA, Ac, Hl);
  scan_pass2<<<(Bn * DH) / 256, 256, 0, stream>>>(Ac, Hl, Hp, h_last);
  scan_pass3<<<(Bn * NCH * DH) / 256, 256, 0, stream>>>(aA, sA, ogA, Hp, ybar);

  // 4) GEMM2 + bias into output.
  gemm2<<<dim3(DM / 64, M / 128), 256, 0, stream>>>(ybar, WoB, bo, y_out);
}
